// TransformerEstimator_5849745457375
// MI455X (gfx1250) — matrix operand order repaired and checked by reading, not compiled, not run on hardware
//
#include <hip/hip_runtime.h>
#include <hip/hip_bf16.h>
#include <math.h>

typedef _Float16 half_t;
typedef _Float16 v8h  __attribute__((ext_vector_type(8)));
typedef _Float16 v16h __attribute__((ext_vector_type(16)));
typedef float    v8f  __attribute__((ext_vector_type(8)));

#define B_  4
#define T_  1000
#define P_  300
#define C_  512
#define D_  256
#define CC_ 256
#define H_  8
#define NL_ 4
#define FF_ 2048
#define L_  1300
#define DH_ 64
#define TP_ 1024          // T padded to 32 (wave row-slab granularity)
#define PP_ 320           // P padded to 32
#define LP_ 1312          // L padded to 32
#define M_  (B_*LP_)      // 5248 = 41*128

// ---------------------------------------------------------------- fragments
__device__ __forceinline__ v16h cat16(v8h lo, v8h hi) {
  return __builtin_shufflevector(lo, hi, 0,1,2,3,4,5,6,7,8,9,10,11,12,13,14,15);
}
// A 16x32 f16 fragment: lane<16 -> K {k0..k0+7, k0+16..k0+23}; lane>=16 shifted by 8
__device__ __forceinline__ v16h ldA(const half_t* row, int k0, int hi) {
  v8h lo = *(const v8h*)(row + k0 + hi*8);
  v8h h2 = *(const v8h*)(row + k0 + 16 + hi*8);
  return cat16(lo, h2);
}
// B 32x16 f16 fragment: lane<16 -> K {k0..k0+7, k0+16..k0+23}; lane>=16 shifted by 8
__device__ __forceinline__ v16h ldB(const half_t* row, int k0, int hi) {
  v8h lo = *(const v8h*)(row + k0 + hi*8);
  v8h h2 = *(const v8h*)(row + k0 + 16 + hi*8);
  return cat16(lo, h2);
}
__device__ __forceinline__ v8f wmma16(v16h a, v16h b, v8f c) {
  return __builtin_amdgcn_wmma_f32_16x16x32_f16(false, a, false, b, (short)0, c, false, false);
}

// ---------------------------------------------------------------- GEMM (TN)
// C[m,n] = sum_k A[m,k]*Bt[n,k] (+bias[n]) (+res[m,n]); A,Bt f16, C f32.
// Workgroup: 128x128 tile, 8 waves (4 row-slabs x 2 col-slabs), wave = 32x64.
// Per k-step: 6 fragment loads feed 8 WMMAs (~21 flop/byte from cache).
__global__ void __launch_bounds__(256)
gemm_tn_kernel(const half_t* __restrict__ A, const half_t* __restrict__ Bt,
               const float* __restrict__ bias, const float* __restrict__ res,
               float* __restrict__ Cm, int M, int N, int K,
               size_t sA, size_t sC)
{
  A  += (size_t)blockIdx.z * sA;
  Cm += (size_t)blockIdx.z * sC;
  if (res) res += (size_t)blockIdx.z * sC;
  int lane = threadIdx.x & 31;
  int wv   = threadIdx.x >> 5;
  int m0 = blockIdx.y*128 + (wv >> 1)*32;
  int n0 = blockIdx.x*128 + (wv & 1)*64;
  if (m0 >= M) return;                 // wave-uniform (M multiple of 32)
  int r16 = lane & 15, hi = lane >> 4;
  const half_t* Ar0 = A + (size_t)(m0 + r16)      * K;
  const half_t* Ar1 = A + (size_t)(m0 + 16 + r16) * K;
  const half_t* Bp[4];
#pragma unroll
  for (int g = 0; g < 4; g++) Bp[g] = Bt + (size_t)(n0 + g*16 + r16) * K;
  v8f acc[2][4];
#pragma unroll
  for (int t = 0; t < 2; t++)
#pragma unroll
    for (int g = 0; g < 4; g++) acc[t][g] = (v8f){};
  for (int k0 = 0; k0 < K; k0 += 32) {
    v16h a0 = ldA(Ar0, k0, hi);
    v16h a1 = ldA(Ar1, k0, hi);
#pragma unroll
    for (int g = 0; g < 4; g++) {
      v16h b = ldB(Bp[g], k0, hi);
      acc[0][g] = wmma16(a0, b, acc[0][g]);
      acc[1][g] = wmma16(a1, b, acc[1][g]);
    }
  }
  float bi[4];
#pragma unroll
  for (int g = 0; g < 4; g++) bi[g] = bias ? bias[n0 + g*16 + r16] : 0.f;
#pragma unroll
  for (int t = 0; t < 2; t++) {
#pragma unroll
    for (int r = 0; r < 8; r++) {
      int row = m0 + t*16 + r + 8*hi;
      size_t base = (size_t)row*N + n0 + r16;
#pragma unroll
      for (int g = 0; g < 4; g++) {
        float v = acc[t][g][r] + bi[g];
        if (res) v += res[base + g*16];
        Cm[base + g*16] = v;
      }
    }
  }
}

// ------------------------------------------------------- flash attention
// q16,k16: (B,H,LP,DH) f16 (q pre-scaled by 1/8); vT16: (B,H,DH,LP) f16
// o16: (B,LP,C) f16. One 16-row Q tile per wave; 32-key tiles.
__global__ void __launch_bounds__(256)
attn_kernel(const half_t* __restrict__ q16, const half_t* __restrict__ k16,
            const half_t* __restrict__ vT16, const unsigned char* __restrict__ keypad,
            half_t* __restrict__ o16)
{
  __shared__ __align__(16) half_t pbuf[8][16*32];
  int lane = threadIdx.x & 31;
  int w    = threadIdx.x >> 5;
  int b = blockIdx.z, h = blockIdx.y;
  int q0 = (blockIdx.x * 8 + w) * 16;
  if (q0 >= LP_) return;               // wave-uniform
  int r16 = lane & 15, hi = lane >> 4;
  const half_t* Qrow = q16 + ((size_t)(b*H_ + h)*LP_ + q0 + r16) * DH_;
  v16h qa0 = ldA(Qrow, 0, hi);
  v16h qa1 = ldA(Qrow, 32, hi);
  const half_t* Kb = k16  + (size_t)(b*H_ + h)*LP_*DH_;
  const half_t* Vb = vT16 + (size_t)(b*H_ + h)*DH_*LP_;
  const unsigned char* kp = keypad + (size_t)b*LP_;
  half_t* P = pbuf[w];
  float m_[8], l_[8];
  v8f o0 = {}, o1 = {}, o2 = {}, o3 = {};
#pragma unroll
  for (int r = 0; r < 8; r++) { m_[r] = -1e30f; l_[r] = 0.f; }

  for (int kt = 0; kt < LP_; kt += 32) {
    v8f s0 = {}, s1 = {};
    const half_t* K0 = Kb + (size_t)(kt + r16) * DH_;
    const half_t* K1 = Kb + (size_t)(kt + 16 + r16) * DH_;
    v16h kbf;
    kbf = ldB(K0, 0, hi);  s0 = wmma16(qa0, kbf, s0);
    kbf = ldB(K0, 32, hi); s0 = wmma16(qa1, kbf, s0);
    kbf = ldB(K1, 0, hi);  s1 = wmma16(qa0, kbf, s1);
    kbf = ldB(K1, 32, hi); s1 = wmma16(qa1, kbf, s1);
    bool pad0 = kp[kt + r16] != 0;
    bool pad1 = kp[kt + 16 + r16] != 0;
#pragma unroll
    for (int r = 0; r < 8; r++) {
      float a0 = pad0 ? -1e9f : s0[r];
      float a1 = pad1 ? -1e9f : s1[r];
      float v = fmaxf(a0, a1);
      v = fmaxf(v, __shfl_xor(v, 1));
      v = fmaxf(v, __shfl_xor(v, 2));
      v = fmaxf(v, __shfl_xor(v, 4));
      v = fmaxf(v, __shfl_xor(v, 8));
      float nm = fmaxf(m_[r], v);
      float alpha = __expf(m_[r] - nm);
      m_[r] = nm;
      float p0 = __expf(a0 - nm);
      float p1 = __expf(a1 - nm);
      float rs = p0 + p1;
      rs += __shfl_xor(rs, 1); rs += __shfl_xor(rs, 2);
      rs += __shfl_xor(rs, 4); rs += __shfl_xor(rs, 8);
      l_[r] = l_[r]*alpha + rs;
      int m = r + 8*hi;
      P[m*32 + r16]      = (half_t)p0;
      P[m*32 + 16 + r16] = (half_t)p1;
      o0[r] *= alpha; o1[r] *= alpha; o2[r] *= alpha; o3[r] *= alpha;
    }
    asm volatile("s_wait_dscnt 0" ::: "memory");   // P stores visible to wave
    v16h pa = ldA(P + r16*32, 0, hi);              // P as 16x32 A fragment
    v16h vb;
    vb = ldB(Vb + (size_t)(r16)      * LP_, kt, hi); o0 = wmma16(pa, vb, o0);
    vb = ldB(Vb + (size_t)(16 + r16) * LP_, kt, hi); o1 = wmma16(pa, vb, o1);
    vb = ldB(Vb + (size_t)(32 + r16) * LP_, kt, hi); o2 = wmma16(pa, vb, o2);
    vb = ldB(Vb + (size_t)(48 + r16) * LP_, kt, hi); o3 = wmma16(pa, vb, o3);
  }
  half_t* Ob = o16 + (size_t)b*LP_*C_ + (size_t)h*DH_;
#pragma unroll
  for (int r = 0; r < 8; r++) {
    int row = q0 + r + 8*hi;
    float inv = 1.f / l_[r];
    half_t* orow = Ob + (size_t)row*C_;
    orow[ 0 + r16] = (half_t)(o0[r]*inv);
    orow[16 + r16] = (half_t)(o1[r]*inv);
    orow[32 + r16] = (half_t)(o2[r]*inv);
    orow[48 + r16] = (half_t)(o3[r]*inv);
  }
}

// ---------------------------------------------------------------- LayerNorm
__global__ void __launch_bounds__(256)
ln_kernel(const float* __restrict__ x, const float* __restrict__ g,
          const float* __restrict__ bb, half_t* __restrict__ out, int M)
{
  int lane = threadIdx.x & 31;
  int w    = threadIdx.x >> 5;
  int row  = blockIdx.x*8 + w;
  if (row >= M) return;
  const float* xr = x + (size_t)row*C_;
  float vals[16]; float s = 0.f;
#pragma unroll
  for (int i = 0; i < 16; i++) { vals[i] = xr[lane + 32*i]; s += vals[i]; }
#pragma unroll
  for (int off = 1; off < 32; off <<= 1) s += __shfl_xor(s, off);
  float mean = s * (1.f/C_);
  float vs = 0.f;
#pragma unroll
  for (int i = 0; i < 16; i++) { float d = vals[i]-mean; vs += d*d; }
#pragma unroll
  for (int off = 1; off < 32; off <<= 1) vs += __shfl_xor(vs, off);
  float rs = rsqrtf(vs * (1.f/C_) + 1e-5f);
  half_t* orow = out + (size_t)row*C_;
#pragma unroll
  for (int i = 0; i < 16; i++) {
    int c = lane + 32*i;
    orow[c] = (half_t)((vals[i]-mean)*rs*g[c] + bb[c]);
  }
}

// ---------------------------------------------------------------- elementwise
__global__ void f2h_kernel(const float* __restrict__ in, half_t* __restrict__ out,
                           size_t n, int relu, float scale) {
  size_t i = (size_t)blockIdx.x*256 + threadIdx.x;
  if (i >= n) return;
  float v = in[i]*scale;
  if (relu) v = fmaxf(v, 0.f);
  out[i] = (half_t)v;
}
__global__ void addscale_kernel(float* __restrict__ x, const float* __restrict__ s, size_t n) {
  size_t i = (size_t)blockIdx.x*256 + threadIdx.x;
  if (i < n) x[i] = (x[i] + s[i]) * 0.70710678118654752f;
}
__global__ void accum_kernel(float* __restrict__ a, const float* __restrict__ x, size_t n) {
  size_t i = (size_t)blockIdx.x*256 + threadIdx.x;
  if (i < n) a[i] += x[i];
}
__global__ void copy_kernel(float* __restrict__ d, const float* __restrict__ s, size_t n) {
  size_t i = (size_t)blockIdx.x*256 + threadIdx.x;
  if (i < n) d[i] = s[i];
}
// W (K x N) f32  ->  Wt (N x K) f16 (output-major, coalesced writes)
__global__ void wtconv_kernel(const float* __restrict__ W, half_t* __restrict__ Wt,
                              int K, int N) {
  size_t i = (size_t)blockIdx.x*256 + threadIdx.x;
  if (i >= (size_t)N*K) return;
  int n = (int)(i / K), k = (int)(i % K);
  Wt[i] = (half_t)W[(size_t)k*N + n];
}
// spec/cond (B, Drows, Tlen) f32 -> (B, TP, Drows) f16, zero-padded rows
__global__ void spect_kernel(const float* __restrict__ in, half_t* __restrict__ out,
                             int Drows, int Tlen) {
  size_t i = (size_t)blockIdx.x*256 + threadIdx.x;
  size_t tot = (size_t)B_*TP_*Drows;
  if (i >= tot) return;
  int d = (int)(i % Drows);
  size_t r = i / Drows;
  int t = (int)(r % TP_);
  int b = (int)(r / TP_);
  out[i] = (t < Tlen) ? (half_t)in[((size_t)b*Drows + d)*Tlen + t] : (half_t)0.f;
}
// prompt (B,P,D) f32 -> (B,PP,D) f16 zero-padded
__global__ void promptc_kernel(const float* __restrict__ in, half_t* __restrict__ out) {
  size_t i = (size_t)blockIdx.x*256 + threadIdx.x;
  size_t tot = (size_t)B_*PP_*D_;
  if (i >= tot) return;
  int d = (int)(i % D_);
  size_t r = i / D_;
  int p = (int)(r % PP_);
  int b = (int)(r / PP_);
  out[i] = (p < P_) ? (half_t)in[((size_t)b*P_ + p)*D_ + d] : (half_t)0.f;
}
__global__ void lens_kernel(const unsigned char* __restrict__ x_mask,
                            const unsigned char* __restrict__ prompt_mask,
                            int* __restrict__ pls, int* __restrict__ sls) {
  int b = threadIdx.x;
  if (b >= B_) return;
  int pl = 0;
  for (int j = 0; j < P_; j++) pl += prompt_mask[b*P_ + j] ? 0 : 1;
  int sl = 0;
  for (int j = 0; j < T_; j++) sl += x_mask[b*T_ + j] ? 0 : 1;
  pls[b] = pl; sls[b] = sl;
}
__global__ void keypad_kernel(const int* __restrict__ pls, const int* __restrict__ sls,
                              unsigned char* __restrict__ kp) {
  int i = blockIdx.x*256 + threadIdx.x;
  if (i >= B_*LP_) return;
  int b = i / LP_, j = i % LP_;
  kp[i] = (j >= pls[b] + sls[b]) ? 1 : 0;
}
// diffusion-step MLP (mish): grid (2,B): kind0 -> dstep(t=step[b]), kind1 -> pstep(t=0)
__global__ void __launch_bounds__(256)
stepmlp_kernel(const float* __restrict__ steps,
               const float* __restrict__ W1, const float* __restrict__ b1,
               const float* __restrict__ W2, const float* __restrict__ b2,
               float* __restrict__ dout, float* __restrict__ pout) {
  __shared__ float e[256];
  __shared__ float hb[1024];
  int kind = blockIdx.x, b = blockIdx.y;
  float t = (kind == 0) ? steps[b] : 0.f;
  int tid = threadIdx.x;
  if (tid < 128) {
    float fr = __expf(-(float)tid * (logf(10000.f) / 127.f));
    float a = t * fr;
    e[tid] = sinf(a); e[tid + 128] = cosf(a);
  }
  __syncthreads();
#pragma unroll
  for (int rep = 0; rep < 4; rep++) {
    int j = tid + rep*256;
    float s = b1[j];
    for (int i = 0; i < 256; i++) s += e[i] * W1[i*1024 + j];
    float sp = (s > 20.f) ? s : log1pf(__expf(s));
    hb[j] = s * tanhf(sp);
  }
  __syncthreads();
  float* out = ((kind == 0) ? dout : pout) + (size_t)b*C_;
#pragma unroll
  for (int rep = 0; rep < 2; rep++) {
    int c = tid + rep*256;
    float s = b2[c];
    for (int i = 0; i < 1024; i++) s += hb[i] * W2[i*512 + c];
    out[c] = s;
  }
}
// pack embeddings -> x (B,LP,C) f32
__global__ void embed_kernel(const float* __restrict__ spec_h, const float* __restrict__ cond_h,
                             const float* __restrict__ prompt_h, const float* __restrict__ dstep,
                             const float* __restrict__ pstep, const int* __restrict__ pls,
                             const int* __restrict__ sls, float* __restrict__ x) {
  size_t i = (size_t)blockIdx.x*256 + threadIdx.x;
  if (i >= (size_t)M_*C_) return;
  int c = (int)(i % C_);
  size_t r = i / C_;
  int j = (int)(r % LP_);
  int b = (int)(r / LP_);
  if (j >= L_) { x[i] = 0.f; return; }
  int pl = pls[b], sl = sls[b];
  float base = 0.f, step = 0.f, cv = 0.f;
  int pos;
  if (j < pl) {
    base = prompt_h[((size_t)b*PP_ + j)*C_ + c];
    step = pstep[(size_t)b*C_ + c];
    pos = j + 1;
  } else if (j < pl + sl) {
    int si = j - pl;
    base = spec_h[((size_t)b*TP_ + si)*C_ + c];
    cv   = cond_h[((size_t)b*TP_ + si)*C_ + c];
    step = dstep[(size_t)b*C_ + c];
    pos = si + 1;
  } else {
    pos = j - pl - sl + 1;
  }
  int fi = (c < 256) ? c : c - 256;
  float fr = __expf(-(float)fi * (logf(10000.f) / 255.f));
  float a = (float)pos * fr;
  float pe = (c < 256) ? sinf(a) : cosf(a);
  x[i] = base + pe + cv + step;
}
// qkv (B*LP, 3C) f32 -> q16,k16 (B,H,LP,DH) f16 (q scaled 1/8), vT16 (B,H,DH,LP) f16
__global__ void splitqkv_kernel(const float* __restrict__ qkv, half_t* __restrict__ q,
                                half_t* __restrict__ k, half_t* __restrict__ vT) {
  size_t i = (size_t)blockIdx.x*256 + threadIdx.x;
  if (i >= (size_t)B_*H_*LP_*DH_) return;
  int d = (int)(i % DH_);
  size_t r = i / DH_;
  int l = (int)(r % LP_);
  size_t r2 = r / LP_;
  int h = (int)(r2 % H_);
  int b = (int)(r2 / H_);
  const float* base = qkv + ((size_t)b*LP_ + l)*(3*C_) + h*DH_ + d;
  q[i] = (half_t)(base[0] * 0.125f);
  k[i] = (half_t)base[C_];
  vT[(((size_t)(b*H_ + h))*DH_ + d)*LP_ + l] = (half_t)base[2*C_];
}
// fin (B*LP, D) f32 -> out (B, D, T) f32 gathered at rows pl..pl+T-1
__global__ void gather_kernel(const float* __restrict__ fin, const int* __restrict__ pls,
                              float* __restrict__ out) {
  size_t i = (size_t)blockIdx.x*256 + threadIdx.x;
  if (i >= (size_t)B_*D_*T_) return;
  int t = (int)(i % T_);
  size_t r = i / T_;
  int d = (int)(r % D_);
  int b = (int)(r / D_);
  out[i] = fin[((size_t)b*LP_ + pls[b] + t)*D_ + d];
}

// ================================================================ launcher
extern "C" void kernel_launch(void* const* d_in, const int* in_sizes, int n_in,
                              void* d_out, int out_size, void* d_ws, size_t ws_size,
                              hipStream_t stream) {
  (void)in_sizes; (void)n_in; (void)out_size; (void)ws_size;
  const float* spec        = (const float*)d_in[0];
  const unsigned char* xm  = (const unsigned char*)d_in[1];
  const float* dstep_in    = (const float*)d_in[2];
  const float* cond        = (const float*)d_in[3];
  const float* prompt      = (const float*)d_in[5];
  const unsigned char* pm  = (const unsigned char*)d_in[6];
  const float* spec_W  = (const float*)d_in[7];
  const float* spec_b  = (const float*)d_in[8];
  const float* cond_W  = (const float*)d_in[9];
  const float* cond_b  = (const float*)d_in[10];
  const float* prompt_W= (const float*)d_in[11];
  const float* prompt_b= (const float*)d_in[12];
  const float* mlp_W1  = (const float*)d_in[13];
  const float* mlp_b1  = (const float*)d_in[14];
  const float* mlp_W2  = (const float*)d_in[15];
  const float* mlp_b2  = (const float*)d_in[16];
  const float* out_W   = (const float*)d_in[17];
  const float* out_b   = (const float*)d_in[18];
  const float* skip_W  = (const float*)d_in[19];
  const float* skip_b  = (const float*)d_in[20];
  const float* Wqkv    = (const float*)d_in[21];
  const float* bqkv    = (const float*)d_in[22];
  const float* Wo      = (const float*)d_in[23];
  const float* bo      = (const float*)d_in[24];
  const float* ln1_g   = (const float*)d_in[25];
  const float* ln1_b   = (const float*)d_in[26];
  const float* ln2_g   = (const float*)d_in[27];
  const float* ln2_b   = (const float*)d_in[28];
  const float* ffn_W1  = (const float*)d_in[29];
  const float* ffn_b1  = (const float*)d_in[30];
  const float* ffn_W2  = (const float*)d_in[31];
  const float* ffn_b2  = (const float*)d_in[32];
  float* dout = (float*)d_out;

  char* ws = (char*)d_ws;
  size_t off = 0;
  auto alloc = [&](size_t bytes) -> void* {
    off = (off + 255) & ~(size_t)255;
    void* p = ws + off;
    off += bytes;
    return p;
  };
  // f16 buffers
  half_t* wt_spec   = (half_t*)alloc(sizeof(half_t)*(size_t)C_*D_);
  half_t* wt_cond   = (half_t*)alloc(sizeof(half_t)*(size_t)C_*CC_);
  half_t* wt_prompt = (half_t*)alloc(sizeof(half_t)*(size_t)C_*D_);
  half_t* wt_skip   = (half_t*)alloc(sizeof(half_t)*(size_t)C_*C_);
  half_t* wt_out    = (half_t*)alloc(sizeof(half_t)*(size_t)D_*C_);
  half_t* wt_qkv    = (half_t*)alloc(sizeof(half_t)*(size_t)NL_*3*C_*C_);
  half_t* wt_o      = (half_t*)alloc(sizeof(half_t)*(size_t)NL_*C_*C_);
  half_t* wt_ff1    = (half_t*)alloc(sizeof(half_t)*(size_t)NL_*FF_*C_);
  half_t* wt_ff2    = (half_t*)alloc(sizeof(half_t)*(size_t)NL_*C_*FF_);
  half_t* spec_t    = (half_t*)alloc(sizeof(half_t)*(size_t)B_*TP_*D_);
  half_t* cond_t    = (half_t*)alloc(sizeof(half_t)*(size_t)B_*TP_*CC_);
  half_t* prompt16  = (half_t*)alloc(sizeof(half_t)*(size_t)B_*PP_*D_);
  half_t* h16       = (half_t*)alloc(sizeof(half_t)*(size_t)M_*C_);
  half_t* q16       = (half_t*)alloc(sizeof(half_t)*(size_t)M_*C_);
  half_t* k16       = (half_t*)alloc(sizeof(half_t)*(size_t)M_*C_);
  half_t* vT16      = (half_t*)alloc(sizeof(half_t)*(size_t)M_*C_);
  half_t* o16       = (half_t*)alloc(sizeof(half_t)*(size_t)M_*C_);
  half_t* ff16      = (half_t*)alloc(sizeof(half_t)*(size_t)M_*FF_);
  half_t* y16       = (half_t*)alloc(sizeof(half_t)*(size_t)M_*C_);
  // f32 buffers
  float* spec_h   = (float*)alloc(sizeof(float)*(size_t)B_*TP_*C_);
  float* cond_h   = (float*)alloc(sizeof(float)*(size_t)B_*TP_*C_);
  float* prompt_h = (float*)alloc(sizeof(float)*(size_t)B_*PP_*C_);
  float* dstep    = (float*)alloc(sizeof(float)*(size_t)B_*C_);
  float* pstep    = (float*)alloc(sizeof(float)*(size_t)B_*C_);
  float* x        = (float*)alloc(sizeof(float)*(size_t)M_*C_);
  float* saved0   = (float*)alloc(sizeof(float)*(size_t)M_*C_);
  float* saved1   = (float*)alloc(sizeof(float)*(size_t)M_*C_);
  float* ffn_sum  = (float*)alloc(sizeof(float)*(size_t)M_*C_);
  float* tmp      = (float*)alloc(sizeof(float)*(size_t)M_*C_);
  float* qkv      = (float*)alloc(sizeof(float)*(size_t)M_*3*C_);
  float* ff       = (float*)alloc(sizeof(float)*(size_t)M_*FF_);
  float* fin      = (float*)alloc(sizeof(float)*(size_t)M_*D_);
  int*   pls      = (int*)alloc(sizeof(int)*B_);
  int*   sls      = (int*)alloc(sizeof(int)*B_);
  unsigned char* kp = (unsigned char*)alloc((size_t)B_*LP_);

  auto nb = [](size_t n) { return dim3((unsigned)((n + 255)/256)); };
  auto gemm = [&](const half_t* A, const half_t* Bt, const float* bias, const float* res,
                  float* Co, int M, int N, int K, int batch, size_t sA, size_t sC) {
    dim3 g((unsigned)(N/128), (unsigned)((M + 127)/128), (unsigned)batch);
    gemm_tn_kernel<<<g, 256, 0, stream>>>(A, Bt, bias, res, Co, M, N, K, sA, sC);
  };
  auto wconv = [&](const float* W, half_t* Wt, int K, int N) {
    wtconv_kernel<<<nb((size_t)K*N), 256, 0, stream>>>(W, Wt, K, N);
  };

  // ---- setup
  lens_kernel<<<1, 32, 0, stream>>>(xm, pm, pls, sls);
  keypad_kernel<<<nb((size_t)B_*LP_), 256, 0, stream>>>(pls, sls, kp);
  wconv(spec_W,   wt_spec,   D_, C_);
  wconv(cond_W,   wt_cond,   CC_, C_);
  wconv(prompt_W, wt_prompt, D_, C_);
  wconv(skip_W,   wt_skip,   C_, C_);
  wconv(out_W,    wt_out,    C_, D_);
  for (int i = 0; i < NL_; i++) {
    wconv(Wqkv   + (size_t)i*C_*3*C_, wt_qkv + (size_t)i*3*C_*C_, C_, 3*C_);
    wconv(Wo     + (size_t)i*C_*C_,   wt_o   + (size_t)i*C_*C_,   C_, C_);
    wconv(ffn_W1 + (size_t)i*C_*FF_,  wt_ff1 + (size_t)i*FF_*C_,  C_, FF_);
    wconv(ffn_W2 + (size_t)i*FF_*C_,  wt_ff2 + (size_t)i*C_*FF_,  FF_, C_);
  }
  spect_kernel<<<nb((size_t)B_*TP_*D_),  256, 0, stream>>>(spec, spec_t, D_, T_);
  spect_kernel<<<nb((size_t)B_*TP_*CC_), 256, 0, stream>>>(cond, cond_t, CC_, T_);
  promptc_kernel<<<nb((size_t)B_*PP_*D_), 256, 0, stream>>>(prompt, prompt16);

  // ---- input projections (batched over B)
  gemm(spec_t,   wt_spec,   spec_b,   nullptr, spec_h,   TP_, C_, D_,  B_, (size_t)TP_*D_,  (size_t)TP_*C_);
  gemm(cond_t,   wt_cond,   cond_b,   nullptr, cond_h,   TP_, C_, CC_, B_, (size_t)TP_*CC_, (size_t)TP_*C_);
  gemm(prompt16, wt_prompt, prompt_b, nullptr, prompt_h, PP_, C_, D_,  B_, (size_t)PP_*D_,  (size_t)PP_*C_);
  stepmlp_kernel<<<dim3(2, B_), 256, 0, stream>>>(dstep_in, mlp_W1, mlp_b1, mlp_W2, mlp_b2, dstep, pstep);
  embed_kernel<<<nb((size_t)M_*C_), 256, 0, stream>>>(spec_h, cond_h, prompt_h, dstep, pstep, pls, sls, x);

  const size_t nMC = (size_t)M_*C_;
  dim3 ga((unsigned)((LP_/16 + 7)/8), H_, B_);
  for (int i = 0; i < NL_; i++) {
    if (i == 2) addscale_kernel<<<nb(nMC), 256, 0, stream>>>(x, saved1, nMC);
    if (i == 3) addscale_kernel<<<nb(nMC), 256, 0, stream>>>(x, saved0, nMC);
    ln_kernel<<<(M_ + 7)/8, 256, 0, stream>>>(x, ln1_g + i*C_, ln1_b + i*C_, h16, M_);
    gemm(h16, wt_qkv + (size_t)i*3*C_*C_, bqkv + i*3*C_, nullptr, qkv, M_, 3*C_, C_, 1, 0, 0);
    splitqkv_kernel<<<nb(nMC), 256, 0, stream>>>(qkv, q16, k16, vT16);
    attn_kernel<<<ga, 256, 0, stream>>>(q16, k16, vT16, kp, o16);
    gemm(o16, wt_o + (size_t)i*C_*C_, bo + i*C_, x, x, M_, C_, C_, 1, 0, 0);
    ln_kernel<<<(M_ + 7)/8, 256, 0, stream>>>(x, ln2_g + i*C_, ln2_b + i*C_, h16, M_);
    gemm(h16, wt_ff1 + (size_t)i*FF_*C_, ffn_b1 + i*FF_, nullptr, ff, M_, FF_, C_, 1, 0, 0);
    f2h_kernel<<<nb((size_t)M_*FF_), 256, 0, stream>>>(ff, ff16, (size_t)M_*FF_, 1, 1.f);
    gemm(ff16, wt_ff2 + (size_t)i*C_*FF_, ffn_b2 + i*C_, x, x, M_, C_, FF_, 1, 0, 0);
    if (i == 0) {
      copy_kernel<<<nb(nMC), 256, 0, stream>>>(saved0, x, nMC);
      copy_kernel<<<nb(nMC), 256, 0, stream>>>(ffn_sum, x, nMC);
    } else {
      if (i == 1) copy_kernel<<<nb(nMC), 256, 0, stream>>>(saved1, x, nMC);
      accum_kernel<<<nb(nMC), 256, 0, stream>>>(ffn_sum, x, nMC);
    }
  }
  // ---- head: y = sum/sqrt(NL); relu(y@skip_W+skip_b) @ out_W + out_b; gather
  f2h_kernel<<<nb(nMC), 256, 0, stream>>>(ffn_sum, y16, nMC, 0, 0.5f);
  gemm(y16, wt_skip, skip_b, nullptr, tmp, M_, C_, C_, 1, 0, 0);
  f2h_kernel<<<nb(nMC), 256, 0, stream>>>(tmp, h16, nMC, 1, 1.f);
  gemm(h16, wt_out, out_b, nullptr, fin, M_, D_, C_, 1, 0, 0);
  gather_kernel<<<nb((size_t)B_*D_*T_), 256, 0, stream>>>(fin, pls, dout);
}
